// ScaledDotProductWithMapAttentionV2_30262339567909
// MI455X (gfx1250) — compile-verified, matrix operand order checked by reading, not run on hardware
//
#include <hip/hip_runtime.h>
#include <hip/hip_bf16.h>

// ---------------------------------------------------------------------------
// Dual-key MHA for MI455X (gfx1250), wave32 + V_WMMA_F32_16X16X32_BF16 + TDM.
//   B=4, NQ=NK=1024, D_IN=D_OUT=1024, H=16, D_K=D_V=64
// Pipeline:
//   0) cvt:   queries/keys1/keys2/values fp32 -> bf16 (row-major)
//      wtr:   Wq/Wk1/Wk2/Wv/Wo fp32 [K][N] -> bf16 transposed [N][K]
//   1) Q  = queries@Wq+bq                      -> bf16 [B,H,NQ,64]
//   2) Ks = keys1@Wk1 + keys2@Wk2 + (bk1+bk2)  -> bf16 [B,H,NK,64]
//        (q@k1^T + q@k2^T == q@(k1+k2)^T, q shared)
//   3) Vt = (values@Wv+bv)^T                   -> bf16 [B,H,64,NK]
//   4) flash attention: S=Q@Ks^T *scale*attw, mask, online softmax, O+=P@V
//                                              -> bf16 [B*NQ, H*64]
//   5) out = O@Wo + bo                         -> fp32 d_out
// GEMM W tiles are moved Global->LDS by the Tensor Data Mover
// (tensor_load_to_lds, double-buffered, s_wait_tensorcnt).
// ---------------------------------------------------------------------------

typedef __attribute__((ext_vector_type(16))) __bf16 v16bf;
typedef __attribute__((ext_vector_type(8)))  __bf16 bf16x8;
typedef __attribute__((ext_vector_type(8)))  float  v8f;
typedef __attribute__((ext_vector_type(4)))  unsigned int v4u;
typedef __attribute__((ext_vector_type(8)))  int v8i;
typedef __attribute__((ext_vector_type(4)))  int v4i;

union F16x16 { v16bf v; bf16x8 h[2]; };

// A/B fragment for 16x16x32 bf16 WMMA (wave32):
//   lane l (hi=l>>4): K elements {hi*8..hi*8+7} and {hi*8+16..hi*8+23}
// -> two contiguous 16-byte chunks from the lane's 32-element K window.
__device__ __forceinline__ v16bf load_frag(const __bf16* win, int hi) {
  F16x16 f;
  f.h[0] = *(const bf16x8*)(win + hi * 8);
  f.h[1] = *(const bf16x8*)(win + hi * 8 + 16);
  return f.v;
}

__device__ __forceinline__ v8f wmma_bf16(v16bf a, v16bf b, v8f c) {
  return __builtin_amdgcn_wmma_f32_16x16x32_bf16(false, a, false, b,
                                                 (short)0, c, false, false);
}

__device__ __forceinline__ v8f zero_v8f() {
  v8f z = {0.f, 0.f, 0.f, 0.f, 0.f, 0.f, 0.f, 0.f};
  return z;
}

// ---------------------------------------------------------------------------
// TDM: load a 64x64 bf16 tile (rows stride 1024 elements, row-contiguous)
// from global into LDS. D# built per CDNA5 ISA.
// This toolchain exposes the 6-arg builtin:
//   (uint32x4 g0, int32x8 g1, int32x4 g2, int32x4 g3, int32x8 extra, i32 cpol)
// ---------------------------------------------------------------------------
__device__ __forceinline__ void tdm_load_tile_64x64_bf16(const __bf16* gsrc,
                                                         unsigned lds_byte) {
  const unsigned long long ga = (unsigned long long)(uintptr_t)gsrc;
  v4u g0;
  g0[0] = 1u;                                   // count=1 (valid), user mode
  g0[1] = lds_byte;                             // lds_addr
  g0[2] = (unsigned)(ga & 0xFFFFFFFFu);         // global_addr[31:0]
  g0[3] = (unsigned)((ga >> 32) & 0x01FFFFFFu)  // global_addr[56:32]
          | (2u << 30);                         // type=2 ("image")
  v8i g1;
  g1[0] = (int)(1u << 16);          // workgroup_mask=0, data_size=1 (2 bytes)
  g1[1] = (int)(1024u << 16);       // tensor_dim0[15:0] -> [31:16]
  g1[2] = (int)(1024u << 16);       // tensor_dim0 hi=0 | tensor_dim1[15:0]
  g1[3] = (int)(64u << 16);         // tensor_dim1 hi=0 | tile_dim0=64
  g1[4] = (int)64;                  // tile_dim1=64, tile_dim2=0
  g1[5] = (int)1024;                // tensor_dim0_stride = 1024 elements
  g1[6] = 0;                        // stride0 hi | tensor_dim1_stride lo (2D)
  g1[7] = 0;
  const v4i z4 = {0, 0, 0, 0};      // groups 2/3 unused (<=2D tensor)
  const v8i z8 = {0, 0, 0, 0, 0, 0, 0, 0};
  __builtin_amdgcn_tensor_load_to_lds(g0, g1, z4, z4, z8, 0);
}

__device__ __forceinline__ unsigned lds_offset(const void* p) {
  return (unsigned)(uintptr_t)p;    // generic LDS address: offset in [31:0]
}

enum { EPI_BHND = 0, EPI_BHDN = 1, EPI_F32 = 2 };

// ---------------------------------------------------------------------------
// bf16 GEMM:  C = A@W^T_t (+ A2@W2^T_t) + bias (+bias2)
//   A:  bf16 [4096,1024] row-major (pre-converted)
//   Wt: bf16 [1024,1024] = W transposed, so B-frags are row-contiguous.
// Block: 256 threads = 8 waves; tile 128(M) x 64(N); K-step 64.
// A-frags straight from global (rows are wave-private, L2-resident);
// W tiles streamed into LDS by the TDM, double-buffered one step ahead.
// ---------------------------------------------------------------------------
template<int EPI, bool DUAL>
__launch_bounds__(256)
__global__ void gemm_kernel(const __bf16* __restrict__ A,
                            const __bf16* __restrict__ Wt,
                            const float* __restrict__ bias,
                            const __bf16* __restrict__ A2,
                            const __bf16* __restrict__ Wt2,
                            const float* __restrict__ bias2,
                            void* __restrict__ Cout) {
  constexpr int N = 1024, K = 1024;
  constexpr int NW = DUAL ? 2 : 1;
  __shared__ __align__(16) __bf16 Wl[2][NW][64][64];   // 16/32 KB

  const int tid = threadIdx.x;
  const int lane = tid & 31, wave = tid >> 5;
  const int hi = lane >> 4, ln = lane & 15;
  const int rbase = blockIdx.y * 128;
  const int cbase = blockIdx.x * 64;

  const __bf16* arow = A + (size_t)(rbase + wave * 16 + ln) * K;
  const __bf16* a2row = DUAL ? (A2 + (size_t)(rbase + wave * 16 + ln) * K)
                             : (const __bf16*)nullptr;
  const __bf16* wpan = Wt + (size_t)cbase * K;
  const __bf16* w2pan = DUAL ? (Wt2 + (size_t)cbase * K)
                             : (const __bf16*)nullptr;

  v8f acc[4];
#pragma unroll
  for (int nt = 0; nt < 4; ++nt) acc[nt] = zero_v8f();

  if (wave == 0) {                       // prime buffer 0
    tdm_load_tile_64x64_bf16(wpan, lds_offset(&Wl[0][0][0][0]));
    if constexpr (DUAL)
      tdm_load_tile_64x64_bf16(w2pan, lds_offset(&Wl[0][1][0][0]));
  }

  for (int kt = 0; kt < 16; ++kt) {
    const int k0 = kt * 64;
    const int cur = kt & 1, nxt = cur ^ 1;
    if (wave == 0) {
      if (kt < 15) {                     // prefetch next tile via TDM
        tdm_load_tile_64x64_bf16(wpan + k0 + 64,
                                 lds_offset(&Wl[nxt][0][0][0]));
        if constexpr (DUAL)
          tdm_load_tile_64x64_bf16(w2pan + k0 + 64,
                                   lds_offset(&Wl[nxt][1][0][0]));
        if constexpr (DUAL) __builtin_amdgcn_s_wait_tensorcnt(2);
        else                __builtin_amdgcn_s_wait_tensorcnt(1);
      } else {
        __builtin_amdgcn_s_wait_tensorcnt(0);
      }
    }
    __syncthreads();                     // tile 'cur' visible to all waves

    const v16bf a0 = load_frag(arow + k0, hi);
    const v16bf a1 = load_frag(arow + k0 + 32, hi);
#pragma unroll
    for (int nt = 0; nt < 4; ++nt) {
      const __bf16* wr = &Wl[cur][0][nt * 16 + ln][0];
      acc[nt] = wmma_bf16(a0, load_frag(wr, hi), acc[nt]);
      acc[nt] = wmma_bf16(a1, load_frag(wr + 32, hi), acc[nt]);
    }
    if constexpr (DUAL) {
      const v16bf b0 = load_frag(a2row + k0, hi);
      const v16bf b1 = load_frag(a2row + k0 + 32, hi);
#pragma unroll
      for (int nt = 0; nt < 4; ++nt) {
        const __bf16* wr = &Wl[cur][1][nt * 16 + ln][0];
        acc[nt] = wmma_bf16(b0, load_frag(wr, hi), acc[nt]);
        acc[nt] = wmma_bf16(b1, load_frag(wr + 32, hi), acc[nt]);
      }
    }
    // drain this wave's LDS reads before TDM may overwrite buffer 'nxt'
    asm volatile("s_wait_dscnt 0x0" ::: "memory");
    __syncthreads();
  }

  // ---- epilogue: C layout lane l, VGPR j -> row j+8*(l>=16), col l&15 ----
#pragma unroll
  for (int nt = 0; nt < 4; ++nt) {
    const int cg = cbase + nt * 16 + ln;
    float bsum = bias[cg];
    if constexpr (DUAL) bsum += bias2[cg];
#pragma unroll
    for (int j = 0; j < 8; ++j) {
      const int rg = rbase + wave * 16 + j + 8 * hi;  // 0..4095 = b*1024+n
      const float val = acc[nt][j] + bsum;
      if constexpr (EPI == EPI_F32) {
        ((float*)Cout)[(size_t)rg * N + cg] = val;
      } else {
        const int b = rg >> 10, n = rg & 1023, h = cg >> 6, d = cg & 63;
        if constexpr (EPI == EPI_BHND)
          ((__bf16*)Cout)[(((size_t)b * 16 + h) * 1024 + n) * 64 + d] = (__bf16)val;
        else  // EPI_BHDN (transposed, for V)
          ((__bf16*)Cout)[(((size_t)b * 16 + h) * 64 + d) * 1024 + n] = (__bf16)val;
      }
    }
  }
}

// ---------------------------------------------------------------------------
// fp32 -> bf16 conversion (activations), 8 elements/thread.
// ---------------------------------------------------------------------------
__launch_bounds__(256)
__global__ void cvt_kernel(const float* __restrict__ s, __bf16* __restrict__ d) {
  const size_t i = ((size_t)blockIdx.x * 256 + threadIdx.x) * 8;
  const float4 v0 = *(const float4*)(s + i);
  const float4 v1 = *(const float4*)(s + i + 4);
  bf16x8 o;
  o[0] = (__bf16)v0.x; o[1] = (__bf16)v0.y; o[2] = (__bf16)v0.z; o[3] = (__bf16)v0.w;
  o[4] = (__bf16)v1.x; o[5] = (__bf16)v1.y; o[6] = (__bf16)v1.z; o[7] = (__bf16)v1.w;
  *(bf16x8*)(d + i) = o;
}

// ---------------------------------------------------------------------------
// fp32 [1024][1024] -> bf16 transposed [1024][1024] (LDS-tiled 32x32).
// ---------------------------------------------------------------------------
__launch_bounds__(256)
__global__ void wtrans_kernel(const float* __restrict__ W,
                              __bf16* __restrict__ Wt) {
  __shared__ float t[32][33];
  const int tx = threadIdx.x & 31, ty = threadIdx.x >> 5;  // 32 x 8
  const int rb = blockIdx.y * 32, cb = blockIdx.x * 32;
#pragma unroll
  for (int i = 0; i < 4; ++i)
    t[ty + 8 * i][tx] = W[(size_t)(rb + ty + 8 * i) * 1024 + cb + tx];
  __syncthreads();
#pragma unroll
  for (int i = 0; i < 4; ++i)
    Wt[(size_t)(cb + ty + 8 * i) * 1024 + rb + tx] = (__bf16)t[tx][ty + 8 * i];
}

// ---------------------------------------------------------------------------
// Flash attention with fused weights*scale and mask, online softmax.
// Block: 128 threads = 4 waves; one block = 64 q-rows of one (b,h).
// ---------------------------------------------------------------------------
__launch_bounds__(128)
__global__ void attention_kernel(const __bf16* __restrict__ Q,
                                 const __bf16* __restrict__ Kmat,
                                 const __bf16* __restrict__ Vt,
                                 const float* __restrict__ attw,
                                 const unsigned char* __restrict__ mask,
                                 __bf16* __restrict__ Oout) {
  constexpr int NQ = 1024, NK = 1024;
  __shared__ __align__(16) __bf16 Pl[4][16][64];   // per-wave P tiles

  const int lane = threadIdx.x & 31, wave = threadIdx.x >> 5;
  const int hi = lane >> 4, ln = lane & 15;
  const int bh = blockIdx.x >> 4;       // 0..63  (= b*16 + h)
  const int qt = blockIdx.x & 15;
  const int b = bh >> 4, h = bh & 15;
  const int q0 = qt * 64 + wave * 16;

  const __bf16* Qb = Q    + (size_t)bh * NQ * 64;
  const __bf16* Kb = Kmat + (size_t)bh * NK * 64;
  const __bf16* Vb = Vt   + (size_t)bh * 64 * NK;
  const float* wb = attw + (size_t)bh * NQ * NK;
  const unsigned char* mb = mask + (size_t)bh * NQ * NK;

  const float scale = 0.08838834764831845f;   // 1/sqrt(2*D_K)

  const v16bf aq0 = load_frag(Qb + (size_t)(q0 + ln) * 64, hi);
  const v16bf aq1 = load_frag(Qb + (size_t)(q0 + ln) * 64 + 32, hi);

  v8f o[4];
#pragma unroll
  for (int dt = 0; dt < 4; ++dt) o[dt] = zero_v8f();
  float mrun[8], lrun[8];
#pragma unroll
  for (int j = 0; j < 8; ++j) { mrun[j] = -1e30f; lrun[j] = 0.f; }

  for (int kt = 0; kt < 16; ++kt) {
    const int kbase = kt * 64;

    // ---- S = Q @ Ksum^T : B-frag lane col = k-row of K (natural layout) ----
    v8f s[4];
#pragma unroll
    for (int nt = 0; nt < 4; ++nt) {
      s[nt] = zero_v8f();
      const __bf16* kr = Kb + (size_t)(kbase + nt * 16 + ln) * 64;
      s[nt] = wmma_bf16(aq0, load_frag(kr, hi), s[nt]);
      s[nt] = wmma_bf16(aq1, load_frag(kr + 32, hi), s[nt]);
    }

    if (kt < 15)  // speculative prefetch of next weight tile (global_prefetch)
      __builtin_prefetch(wb + (size_t)(q0 + ln) * NK + kbase + 64, 0, 0);

    // ---- fused scale * attention_weights, mask -> att; row max ----
    float rmax[8];
#pragma unroll
    for (int j = 0; j < 8; ++j) rmax[j] = -1e30f;
#pragma unroll
    for (int nt = 0; nt < 4; ++nt) {
#pragma unroll
      for (int j = 0; j < 8; ++j) {
        const size_t idx =
            (size_t)(q0 + j + 8 * hi) * NK + kbase + nt * 16 + ln;
        const float a = mb[idx] ? -1e30f : s[nt][j] * scale * wb[idx];
        s[nt][j] = a;
        rmax[j] = fmaxf(rmax[j], a);
      }
    }
#pragma unroll
    for (int j = 0; j < 8; ++j) {
      float v = rmax[j];
      v = fmaxf(v, __shfl_xor(v, 1, 32));
      v = fmaxf(v, __shfl_xor(v, 2, 32));
      v = fmaxf(v, __shfl_xor(v, 4, 32));
      v = fmaxf(v, __shfl_xor(v, 8, 32));
      rmax[j] = v;
    }
    float corr[8], rsum[8];
#pragma unroll
    for (int j = 0; j < 8; ++j) {
      const float mnew = fmaxf(mrun[j], rmax[j]);
      corr[j] = __expf(mrun[j] - mnew);
      mrun[j] = mnew;
      rsum[j] = 0.f;
    }
#pragma unroll
    for (int nt = 0; nt < 4; ++nt) {
#pragma unroll
      for (int j = 0; j < 8; ++j) {
        const float p = __expf(s[nt][j] - mrun[j]);
        s[nt][j] = p;
        rsum[j] += p;
      }
    }
#pragma unroll
    for (int j = 0; j < 8; ++j) {
      float v = rsum[j];
      v += __shfl_xor(v, 1, 32);
      v += __shfl_xor(v, 2, 32);
      v += __shfl_xor(v, 4, 32);
      v += __shfl_xor(v, 8, 32);
      lrun[j] = lrun[j] * corr[j] + v;
    }
#pragma unroll
    for (int dt = 0; dt < 4; ++dt)
#pragma unroll
      for (int j = 0; j < 8; ++j) o[dt][j] *= corr[j];

    // ---- P (C layout) -> LDS -> A-frag layout ----
    __syncthreads();
#pragma unroll
    for (int nt = 0; nt < 4; ++nt)
#pragma unroll
      for (int j = 0; j < 8; ++j)
        Pl[wave][j + 8 * hi][nt * 16 + ln] = (__bf16)s[nt][j];
    __syncthreads();
    const v16bf pa0 = load_frag(&Pl[wave][ln][0], hi);
    const v16bf pa1 = load_frag(&Pl[wave][ln][32], hi);

    // ---- O += P @ V  (Vt rows = d, contiguous over k) ----
#pragma unroll
    for (int dt = 0; dt < 4; ++dt) {
      const __bf16* vr = Vb + (size_t)(dt * 16 + ln) * NK + kbase;
      o[dt] = wmma_bf16(pa0, load_frag(vr, hi), o[dt]);
      o[dt] = wmma_bf16(pa1, load_frag(vr + 32, hi), o[dt]);
    }
  }

  // ---- normalize and write [B*NQ, H*64] row-major bf16 ----
#pragma unroll
  for (int j = 0; j < 8; ++j) {
    const float inv = 1.f / lrun[j];
    const int q = q0 + j + 8 * hi;
#pragma unroll
    for (int dt = 0; dt < 4; ++dt)
      Oout[((size_t)b * NQ + q) * 1024 + h * 64 + dt * 16 + ln] =
          (__bf16)(o[dt][j] * inv);
  }
}

// ---------------------------------------------------------------------------
extern "C" void kernel_launch(void* const* d_in, const int* in_sizes, int n_in,
                              void* d_out, int out_size, void* d_ws,
                              size_t ws_size, hipStream_t stream) {
  (void)in_sizes; (void)n_in; (void)out_size; (void)ws_size;
  const float* queries = (const float*)d_in[0];
  const float* keys1   = (const float*)d_in[1];
  const float* keys2   = (const float*)d_in[2];
  const float* values  = (const float*)d_in[3];
  const float* attw    = (const float*)d_in[4];
  const unsigned char* mask = (const unsigned char*)d_in[5];  // jnp bool_ = 1B
  const float* Wq = (const float*)d_in[6];
  const float* bq = (const float*)d_in[7];
  const float* Wk1 = (const float*)d_in[8];
  const float* bk1 = (const float*)d_in[9];
  const float* Wk2 = (const float*)d_in[10];
  const float* bk2 = (const float*)d_in[11];
  const float* Wv = (const float*)d_in[12];
  const float* bv = (const float*)d_in[13];
  const float* Wo = (const float*)d_in[14];
  const float* bo = (const float*)d_in[15];

  constexpr size_t ACT = (size_t)4096 * 1024;   // 4M elements
  constexpr size_t WEL = (size_t)1024 * 1024;   // 1M elements
  __bf16* Aq  = (__bf16*)d_ws;
  __bf16* Ak1 = Aq + ACT;
  __bf16* Ak2 = Ak1 + ACT;
  __bf16* Av  = Ak2 + ACT;
  __bf16* Wtq = Av + ACT;
  __bf16* Wtk1 = Wtq + WEL;
  __bf16* Wtk2 = Wtk1 + WEL;
  __bf16* Wtv = Wtk2 + WEL;
  __bf16* Wto = Wtv + WEL;
  __bf16* Qws  = Wto + WEL;
  __bf16* Kws  = Qws + ACT;
  __bf16* Vtws = Kws + ACT;
  __bf16* Ows  = Vtws + ACT;   // total ~74 MB of d_ws

  // 0) precision/layout pre-passes (all operands L2-resident)
  cvt_kernel<<<dim3(2048), dim3(256), 0, stream>>>(queries, Aq);
  cvt_kernel<<<dim3(2048), dim3(256), 0, stream>>>(keys1, Ak1);
  cvt_kernel<<<dim3(2048), dim3(256), 0, stream>>>(keys2, Ak2);
  cvt_kernel<<<dim3(2048), dim3(256), 0, stream>>>(values, Av);
  dim3 gt(32, 32), bt(256);
  wtrans_kernel<<<gt, bt, 0, stream>>>(Wq, Wtq);
  wtrans_kernel<<<gt, bt, 0, stream>>>(Wk1, Wtk1);
  wtrans_kernel<<<gt, bt, 0, stream>>>(Wk2, Wtk2);
  wtrans_kernel<<<gt, bt, 0, stream>>>(Wv, Wtv);
  wtrans_kernel<<<gt, bt, 0, stream>>>(Wo, Wto);

  // 1-3) projections (TDM-fed WMMA GEMMs)
  dim3 gg(16, 32), bg(256);    // N/64 x M/128 tiles
  gemm_kernel<EPI_BHND, false><<<gg, bg, 0, stream>>>(
      Aq, Wtq, bq, nullptr, nullptr, nullptr, Qws);
  gemm_kernel<EPI_BHND, true><<<gg, bg, 0, stream>>>(
      Ak1, Wtk1, bk1, Ak2, Wtk2, bk2, Kws);
  gemm_kernel<EPI_BHDN, false><<<gg, bg, 0, stream>>>(
      Av, Wtv, bv, nullptr, nullptr, nullptr, Vtws);

  // 4) fused flash attention
  attention_kernel<<<dim3(1024), dim3(128), 0, stream>>>(
      Qws, Kws, Vtws, attw, mask, Ows);

  // 5) output projection -> fp32
  gemm_kernel<EPI_F32, false><<<gg, bg, 0, stream>>>(
      Ows, Wto, bo, nullptr, nullptr, nullptr, d_out);
}
